// SimplicialEGNNLayer_43035572306255
// MI455X (gfx1250) — matrix operand order repaired and checked by reading, not compiled, not run on hardware
//
#include <hip/hip_runtime.h>

// ---------------------------------------------------------------------------
// SimplicialEGNNLayer edge-message kernel for MI455X (gfx1250, wave32, WMMA)
//
//   prep: zero out/stats, W1^T -> f16 [128][288] in ws, x_send/x_rec -> f16 tables
//   state[E,288] staged in LDS via global_load_async_to_lds_b128 (f16 tables)
//   h = state @ W1 + b1                (v_wmma_f32_16x16x32_f16, f32 accum)
//   BN over edge dim (two-pass: stats via f32 atomics, then apply; GEMM recomputed
//   in pass 2 -- cheaper than 1.6 GB of h spill/refill through HBM)
//   msg = silu(h_bn);  gate = sigmoid(msg @ W_inf + b_inf)
//   out[idx_r] += msg * gate           (global_atomic_add_f32, L2-resident rows)
// ---------------------------------------------------------------------------

#define HDIM   128
#define DIN    272          // 2*H + 16
#define KPAD   288          // DIN padded to multiple of 32 (9 K-steps)
#define TILE_M 64           // edges per block
#define NWAVES 8            // 256 threads

typedef __attribute__((ext_vector_type(16))) _Float16 v16h;
typedef __attribute__((ext_vector_type(8)))  _Float16 v8h;
typedef __attribute__((ext_vector_type(4)))  _Float16 v4h;
typedef __attribute__((ext_vector_type(8)))  float    v8f;

__device__ __forceinline__ float sigmoid_f(float x) {
    return 1.0f / (1.0f + __expf(-x));
}

// Low 32 bits of a generic pointer into the LDS aperture == hardware LDS address.
__device__ __forceinline__ uint32_t lds_addr_u32(const void* p) {
    return (uint32_t)(uintptr_t)p;
}

// gfx1250 async copy global->LDS, 16 bytes per lane, tracked by ASYNCcnt.
__device__ __forceinline__ void async_copy_b128(uint32_t lds_byte_addr, const void* gaddr) {
    asm volatile("global_load_async_to_lds_b128 %0, %1, off"
                 :: "v"(lds_byte_addr), "v"((uint64_t)(uintptr_t)gaddr)
                 : "memory");
}

__device__ __forceinline__ void wait_asynccnt0() {
#if __has_builtin(__builtin_amdgcn_s_wait_asynccnt)
    __builtin_amdgcn_s_wait_asynccnt(0);
#else
    asm volatile("s_wait_asynccnt 0" ::: "memory");
#endif
}

// ---------------------------------------------------------------------------
// prep: zero output + stats, build f16 W1^T [HDIM][KPAD] (B-fragment = two
// 16B row loads per lane), optionally convert node tables to f16.
// ---------------------------------------------------------------------------
__global__ void prep_kernel(const float* __restrict__ W1,
                            const float* __restrict__ x_send,
                            const float* __restrict__ x_rec,
                            float* __restrict__ out,
                            float* __restrict__ stats,
                            _Float16* __restrict__ wsW,
                            _Float16* __restrict__ wsXs,
                            _Float16* __restrict__ wsXr,
                            int out_n, int conv_n) {
    const int stride = gridDim.x * blockDim.x;
    const int gid = blockIdx.x * blockDim.x + threadIdx.x;
    for (int i = gid; i < out_n; i += stride) out[i] = 0.0f;
    for (int i = gid; i < 2 * HDIM; i += stride) stats[i] = 0.0f;
    for (int i = gid; i < HDIM * KPAD; i += stride) {
        const int n = i / KPAD;
        const int k = i - n * KPAD;
        wsW[i] = (k < DIN) ? (_Float16)W1[k * HDIM + n] : (_Float16)0.0f;
    }
    for (int i = gid; i < conv_n; i += stride) {   // conv_n == 0 if ws too small
        wsXs[i] = (_Float16)x_send[i];
        wsXr[i] = (_Float16)x_rec[i];
    }
}

// ---------------------------------------------------------------------------
// Main edge kernel. PASS==1: accumulate BN statistics.
//                   PASS==2: recompute GEMM, apply BN+SiLU+gate, scatter.
// F16TBL: gather 256B f16 rows via async-to-LDS; else gather f32 + convert.
// ---------------------------------------------------------------------------
template <int PASS, bool F16TBL>
__global__ __launch_bounds__(256) void edge_kernel(
    const float* __restrict__ x_send, const float* __restrict__ x_rec,
    const _Float16* __restrict__ xsTbl, const _Float16* __restrict__ xrTbl,
    const int* __restrict__ idx_send, const int* __restrict__ idx_rec,
    const float* __restrict__ edge_attr,
    const _Float16* __restrict__ wsW,
    const float* __restrict__ b1, const float* __restrict__ gamma,
    const float* __restrict__ beta, const float* __restrict__ w_inf,
    const float* __restrict__ b_inf,
    float* __restrict__ stats, float* __restrict__ out,
    int n_edges) {

    __shared__ __align__(32) _Float16 ldsA[TILE_M * KPAD]; // 36864 B; aliased as msg in pass 2
    __shared__ int   ldsIs[TILE_M];
    __shared__ int   ldsIr[TILE_M];
    __shared__ float ldsWinf[HDIM];
    __shared__ float ldsWgt[TILE_M];

    float* msg_lds = (float*)ldsA;   // 64*128*4 = 32768 B <= 36864 B, reused after barrier

    const int tid  = threadIdx.x;
    const int lane = tid & 31;
    const int wave = tid >> 5;
    const int base = blockIdx.x * TILE_M;
    const int nl   = lane & 15;
    const int half = lane >> 4;

    // --- stage per-edge indices + W_inf ---
    if (tid < TILE_M) {
        const int ge = base + tid;
        ldsIs[tid] = (ge < n_edges) ? idx_send[ge] : 0;
        ldsIr[tid] = (ge < n_edges) ? idx_rec[ge] : 0;
    } else if (tid < TILE_M + HDIM) {
        ldsWinf[tid - TILE_M] = w_inf[tid - TILE_M];
    }
    __syncthreads();

    // --- gather state rows into LDS (wave w owns edges 8w..8w+7) ---
    if constexpr (F16TBL) {
        // Async path: one b128 per lane; lanes 0-15 -> edge m, lanes 16-31 -> edge m+1.
        for (int p = 0; p < 4; ++p) {
            const int m = wave * 8 + p * 2 + half;
            const _Float16* gs = xsTbl + (size_t)ldsIs[m] * HDIM + nl * 8;
            const _Float16* gr = xrTbl + (size_t)ldsIr[m] * HDIM + nl * 8;
            async_copy_b128(lds_addr_u32(&ldsA[m * KPAD + nl * 8]), gs);
            async_copy_b128(lds_addr_u32(&ldsA[m * KPAD + HDIM + nl * 8]), gr);
        }
    } else {
        for (int e = 0; e < 8; ++e) {
            const int m  = wave * 8 + e;
            const int ge = base + m;
            const bool valid = ge < n_edges;
            if (e < 7) {
                __builtin_prefetch(x_send + (size_t)ldsIs[m + 1] * HDIM, 0, 1);
            }
            float4 vs = make_float4(0.f, 0.f, 0.f, 0.f);
            float4 vr = make_float4(0.f, 0.f, 0.f, 0.f);
            if (valid) {
                const float4* ps = (const float4*)(x_send + (size_t)ldsIs[m] * HDIM);
                const float4* pr = (const float4*)(x_rec  + (size_t)ldsIr[m] * HDIM);
                vs = ps[lane];
                vr = pr[lane];
            }
            v4h hs = {(_Float16)vs.x, (_Float16)vs.y, (_Float16)vs.z, (_Float16)vs.w};
            v4h hr = {(_Float16)vr.x, (_Float16)vr.y, (_Float16)vr.z, (_Float16)vr.w};
            *(v4h*)(&ldsA[m * KPAD + lane * 4])        = hs;
            *(v4h*)(&ldsA[m * KPAD + HDIM + lane * 4]) = hr;
        }
    }
    // edge_attr (16 floats/edge): 2 edges per 32 lanes; also zero the K pad
    for (int p = 0; p < 4; ++p) {
        const int m   = wave * 8 + p * 2 + half;
        const int ge  = base + m;
        const float a = (ge < n_edges) ? edge_attr[(size_t)ge * 16 + nl] : 0.0f;
        ldsA[m * KPAD + 2 * HDIM + nl]      = (_Float16)a;
        ldsA[m * KPAD + 2 * HDIM + 16 + nl] = (_Float16)0.0f;
    }
    if constexpr (F16TBL) {
        wait_asynccnt0();   // this wave's async LDS writes have landed
    }
    __syncthreads();

    // --- GEMM: wave -> one 16-feature N tile, 4 M tiles x 9 K steps ---
    v8f accz = {0.f, 0.f, 0.f, 0.f, 0.f, 0.f, 0.f, 0.f};
    v8f acc[4] = {accz, accz, accz, accz};

    const _Float16* Wrow = wsW + (size_t)(wave * 16 + nl) * KPAD;
    for (int kt = 0; kt < KPAD / 32; ++kt) {
        const int k0 = kt * 32;
        // B fragment: lane=N, e<8 -> K=k0+8*half+e ; e>=8 -> +16
        v8h b0 = *(const v8h*)(Wrow + k0 + 8 * half);
        v8h b1v = *(const v8h*)(Wrow + k0 + 8 * half + 16);
        v16h bf = __builtin_shufflevector(b0, b1v, 0, 1, 2, 3, 4, 5, 6, 7,
                                          8, 9, 10, 11, 12, 13, 14, 15);
#pragma unroll
        for (int mt = 0; mt < 4; ++mt) {
            // A fragment per ISA layout: e<8 -> K=k0+8*half+e ; e>=8 -> +16
            const _Float16* Arow = &ldsA[(mt * 16 + nl) * KPAD + k0 + 8 * half];
            v8h a0 = *(const v8h*)(Arow);
            v8h a1 = *(const v8h*)(Arow + 16);
            v16h af = __builtin_shufflevector(a0, a1, 0, 1, 2, 3, 4, 5, 6, 7,
                                              8, 9, 10, 11, 12, 13, 14, 15);
            acc[mt] = __builtin_amdgcn_wmma_f32_16x16x32_f16(
                false, af, false, bf, (short)0, acc[mt], false, false);
        }
    }

    // D layout: element r of acc[mt] is edge row m = mt*16 + r + 8*half, feature f
    const int   f    = wave * 16 + nl;
    const float bias = b1[f];

    if (PASS == 1) {
        float s = 0.f, ss = 0.f;
#pragma unroll
        for (int mt = 0; mt < 4; ++mt) {
#pragma unroll
            for (int r = 0; r < 8; ++r) {
                const int m = mt * 16 + r + 8 * half;
                if (base + m < n_edges) {
                    const float h = acc[mt][r] + bias;
                    s  += h;
                    ss += h * h;
                }
            }
        }
        // lanes l and l+16 hold the same feature: pairwise combine, then atomics
        s  += __shfl_xor(s, 16);
        ss += __shfl_xor(ss, 16);
        if (half == 0) {
            unsafeAtomicAdd(&stats[f], s);
            unsafeAtomicAdd(&stats[HDIM + f], ss);
        }
    } else {
        const float invE = 1.0f / (float)n_edges;
        const float mean = stats[f] * invE;
        const float var  = stats[HDIM + f] * invE - mean * mean;
        const float rstd = __frsqrt_rn(var + 1e-5f);
        const float g    = gamma[f];
        const float be   = beta[f];

        __syncthreads();   // everyone finished reading ldsA; safe to alias as msg
#pragma unroll
        for (int mt = 0; mt < 4; ++mt) {
#pragma unroll
            for (int r = 0; r < 8; ++r) {
                const int   m   = mt * 16 + r + 8 * half;
                const float h   = acc[mt][r] + bias;
                const float hn  = (h - mean) * rstd * g + be;
                const float msg = hn * sigmoid_f(hn);   // SiLU
                msg_lds[m * HDIM + f] = msg;
            }
        }
        __syncthreads();

        // --- per-edge gate: thread t -> edge t>>2, feature quarter t&3 ---
        {
            const int e = tid >> 2;
            const int q = tid & 3;
            float s = 0.f;
            const float* mrow = &msg_lds[e * HDIM + q * 32];
            const float* wrow = &ldsWinf[q * 32];
#pragma unroll
            for (int i = 0; i < 32; ++i) s += mrow[i] * wrow[i];
            s += __shfl_xor(s, 1);
            s += __shfl_xor(s, 2);
            if (q == 0) ldsWgt[e] = sigmoid_f(s + b_inf[0]);
        }
        __syncthreads();

        // --- scatter: out[idx_rec[e]] += msg * gate (f32 global atomics) ---
        for (int e = 0; e < 8; ++e) {
            const int m  = wave * 8 + e;
            const int ge = base + m;
            if (ge < n_edges) {
                const float wgt = ldsWgt[m];
                float* orow = out + (size_t)ldsIr[m] * HDIM;
#pragma unroll
                for (int j = 0; j < 4; ++j) {
                    unsafeAtomicAdd(&orow[lane * 4 + j],
                                    msg_lds[m * HDIM + lane * 4 + j] * wgt);
                }
            }
        }
    }
}

// ---------------------------------------------------------------------------
extern "C" void kernel_launch(void* const* d_in, const int* in_sizes, int n_in,
                              void* d_out, int out_size, void* d_ws, size_t ws_size,
                              hipStream_t stream) {
    const float* x_send    = (const float*)d_in[0];
    const float* x_rec     = (const float*)d_in[1];
    const int*   idx_send  = (const int*)d_in[2];
    const int*   idx_rec   = (const int*)d_in[3];
    const float* edge_attr = (const float*)d_in[4];
    const float* W1        = (const float*)d_in[5];
    const float* b1        = (const float*)d_in[6];
    const float* gamma_p   = (const float*)d_in[7];
    const float* beta_p    = (const float*)d_in[8];
    const float* w_inf     = (const float*)d_in[9];
    const float* b_inf     = (const float*)d_in[10];
    float*       out       = (float*)d_out;
    const int    n_edges   = in_sizes[2];
    const int    node_elems = in_sizes[0];           // n_nodes * HDIM
    (void)n_in;

    // ws layout: [0,1KB) f32 stats | [4KB,...) f16 W1^T | [128KB,...) f16 node tables
    float*    stats = (float*)d_ws;
    _Float16* wsW   = (_Float16*)((char*)d_ws + 4096);
    _Float16* wsXs  = (_Float16*)((char*)d_ws + 131072);
    _Float16* wsXr  = wsXs + (size_t)node_elems;

    const size_t ws_needed = 131072 + 2 * (size_t)node_elems * sizeof(_Float16);
    const bool   fast      = ws_size >= ws_needed;

    prep_kernel<<<8192, 256, 0, stream>>>(W1, x_send, x_rec, out, stats,
                                          wsW, wsXs, wsXr,
                                          out_size, fast ? node_elems : 0);

    const int nblk = (n_edges + TILE_M - 1) / TILE_M;
    if (fast) {
        edge_kernel<1, true><<<nblk, 256, 0, stream>>>(
            x_send, x_rec, wsXs, wsXr, idx_send, idx_rec, edge_attr, wsW,
            b1, gamma_p, beta_p, w_inf, b_inf, stats, out, n_edges);
        edge_kernel<2, true><<<nblk, 256, 0, stream>>>(
            x_send, x_rec, wsXs, wsXr, idx_send, idx_rec, edge_attr, wsW,
            b1, gamma_p, beta_p, w_inf, b_inf, stats, out, n_edges);
    } else {
        edge_kernel<1, false><<<nblk, 256, 0, stream>>>(
            x_send, x_rec, nullptr, nullptr, idx_send, idx_rec, edge_attr, wsW,
            b1, gamma_p, beta_p, w_inf, b_inf, stats, out, n_edges);
        edge_kernel<2, false><<<nblk, 256, 0, stream>>>(
            x_send, x_rec, nullptr, nullptr, idx_send, idx_rec, edge_attr, wsW,
            b1, gamma_p, beta_p, w_inf, b_inf, stats, out, n_edges);
    }
}
